// ManyToManyEncoder_44341242364565
// MI455X (gfx1250) — matrix operand order repaired and checked by reading, not compiled, not run on hardware
//
#include <hip/hip_runtime.h>

// ---------------------------------------------------------------------------
// ManyToManyEncoder for MI455X (gfx1250, wave32, WMMA).
//   out = mask( relu(relu([a|mean_b] @ W_in + b_in) @ W_h + b_h) @ W_out + b_out )
// GEMMs: v_wmma_f32_16x16x32_f16; double-buffered LDS staged with
// global_load_async_to_lds_b128 (ASYNCcnt); wave tile 32x64 (8 WMMA/slab).
// ---------------------------------------------------------------------------

typedef __attribute__((ext_vector_type(16))) _Float16 v16h;
typedef __attribute__((ext_vector_type(8)))  _Float16 h8;
typedef __attribute__((ext_vector_type(8)))  float    v8f;
typedef __attribute__((ext_vector_type(4)))  int      v4i;

#define D_MODEL     512
#define BATCH_A     8192
#define MAX_MATCHES 16
#define D_HID       1024
#define D_IN        1024

#if __has_builtin(__builtin_amdgcn_global_load_async_to_lds_b128)
#define HAS_ASYNC_LDS 1
#else
#define HAS_ASYNC_LDS 0
#endif

typedef __attribute__((address_space(1))) v4i* gv4p;  // global int4*
typedef __attribute__((address_space(3))) v4i* lv4p;  // LDS int4*

// One 16-byte chunk: global -> LDS, async (no VGPR round-trip).
__device__ __forceinline__ void cp16_g2l(const _Float16* g, _Float16* l) {
#if HAS_ASYNC_LDS
  __builtin_amdgcn_global_load_async_to_lds_b128(
      (gv4p)(_Float16*)g, (lv4p)l, /*offset=*/0, /*cpol=*/0);
#else
  *(h8*)l = *(const h8*)g;
#endif
}

__device__ __forceinline__ void wait_async0() {
#if __has_builtin(__builtin_amdgcn_s_wait_asynccnt)
  __builtin_amdgcn_s_wait_asynccnt(0);
#else
  asm volatile("s_wait_asynccnt 0x0" ::: "memory");
#endif
}

// ---------------- weight convert + transpose: f32 W[K][N] -> f16 WT[N][K] ---
__global__ __launch_bounds__(256)
void wt_transpose_f16(const float* __restrict__ W, _Float16* __restrict__ WT,
                      int K, int N) {
  __shared__ float tile[32][33];
  const int n0 = blockIdx.x * 32, k0 = blockIdx.y * 32;
  const int tx = threadIdx.x, ty = threadIdx.y;  // 32 x 8
#pragma unroll
  for (int j = 0; j < 32; j += 8)
    tile[ty + j][tx] = W[(long)(k0 + ty + j) * N + (n0 + tx)];
  __syncthreads();
#pragma unroll
  for (int j = 0; j < 32; j += 8)
    WT[(long)(n0 + ty + j) * K + (k0 + tx)] = (_Float16)tile[tx][ty + j];
}

// ---------------- gather + masked mean + pack to f16 activations ------------
__global__ __launch_bounds__(256)
void gather_mean_pack(const float* __restrict__ A,
                      const float* __restrict__ Bt,
                      const int*   __restrict__ idx,
                      _Float16*    __restrict__ X,
                      int*         __restrict__ valid) {
  const int row = blockIdx.x;
  const int t   = threadIdx.x;  // 256 threads, 2 cols each
  __shared__ int sidx[MAX_MATCHES];
  if (t < MAX_MATCHES) sidx[t] = idx[row * MAX_MATCHES + t];
  __syncthreads();

  int cnt = 0;
#pragma unroll
  for (int j = 0; j < MAX_MATCHES; ++j) cnt += (sidx[j] >= 0) ? 1 : 0;
  const float inv = 1.0f / (float)(cnt > 0 ? cnt : 1);

  const int c = t * 2;
  float ax = 0.f, ay = 0.f;
#pragma unroll
  for (int j = 0; j < MAX_MATCHES; ++j) {
    const int s = sidx[j];
    if (s >= 0) {
      const float2 v = *(const float2*)(Bt + (long)s * D_MODEL + c);
      ax += v.x; ay += v.y;
    }
  }
  const float2 av = *(const float2*)(A + (long)row * D_MODEL + c);
  _Float16* xr = X + (long)row * D_IN;
  xr[c]               = (_Float16)av.x;
  xr[c + 1]           = (_Float16)av.y;
  xr[D_MODEL + c]     = (_Float16)(ax * inv);
  xr[D_MODEL + c + 1] = (_Float16)(ay * inv);
  if (t == 0) valid[row] = (cnt > 0) ? 1 : 0;
}

// ---------------- WMMA GEMM: C[M,N] = act(X[M,K] @ WT[N,K]^T + bias) --------
// 256 threads (8 wave32s) -> 64(M) x 256(N) block tile.
// Waves: 2(M-strips of 32) x 4(N-strips of 64); wave tile 32x64 = 2x4 WMMAs.
// K in slabs of 32, double-buffered LDS, async global->LDS staging.
#define BM   64
#define BN   256
#define KT   32
#define LDSP 48   // padded LDS row stride (halfs); 96 B rows, 16-B aligned

__global__ __launch_bounds__(256, 1)
void gemm_wmma(const _Float16* __restrict__ X,    // [M,K] row-major f16
               const _Float16* __restrict__ WT,   // [N,K] row-major f16
               const float*    __restrict__ bias, // [N] f32
               void*           __restrict__ outp, // f16 [M,N] (mode 0) / f32 (mode 1)
               const int*      __restrict__ valid,
               int M, int N, int K, int mode) {
  __shared__ __align__(16) _Float16 Xs[2][BM][LDSP];  // 12 KB
  __shared__ __align__(16) _Float16 Ws[2][BN][LDSP];  // 48 KB

  const int tid  = threadIdx.x;
  const int lane = tid & 31;        // wave32
  const int w    = tid >> 5;        // 8 waves
  const int wm   = w >> 2;          // 0..1 -> 32-row strip
  const int wn   = w & 3;           // 0..3 -> 64-col strip
  const int lr   = lane & 15;
  const int lg   = lane >> 4;       // half-wave select
  const int m0   = blockIdx.x * BM;
  const int n0   = blockIdx.y * BN;

  // Per-thread staging coordinates (16-B chunks).
  const int xr_ = tid >> 2, xc_ = (tid & 3) * 8;  // X: 256 chunks, 1/thread

  v8f acc[2][4];
#pragma unroll
  for (int i = 0; i < 2; ++i)
#pragma unroll
    for (int j = 0; j < 4; ++j) acc[i][j] = (v8f)0.0f;

  const int nk = K / KT;

  // ---- stage slab 0 into buffer 0 ----
  cp16_g2l(X + (long)(m0 + xr_) * K + xc_, &Xs[0][xr_][xc_]);
#pragma unroll
  for (int i = 0; i < 4; ++i) {                    // W: 1024 chunks, 4/thread
    const int c = tid + 256 * i, r = c >> 2, cc = (c & 3) * 8;
    cp16_g2l(WT + (long)(n0 + r) * K + cc, &Ws[0][r][cc]);
  }
  wait_async0();
  __syncthreads();

  for (int s = 0; s < nk; ++s) {
    const int cur = s & 1;
    const int nxt = cur ^ 1;
    // ---- async-stage next slab while computing this one ----
    if (s + 1 < nk) {
      const int kt = (s + 1) * KT;
      cp16_g2l(X + (long)(m0 + xr_) * K + kt + xc_, &Xs[nxt][xr_][xc_]);
#pragma unroll
      for (int i = 0; i < 4; ++i) {
        const int c = tid + 256 * i, r = c >> 2, cc = (c & 3) * 8;
        cp16_g2l(WT + (long)(n0 + r) * K + kt + cc, &Ws[nxt][r][cc]);
      }
    }

    // ---- A fragments (16x32 f16): lane lr = row; ISA half-wave layout ----
    v16h afrag[2], bfrag[4];
#pragma unroll
    for (int tm = 0; tm < 2; ++tm) {
      const _Float16* p = &Xs[cur][wm * 32 + tm * 16 + lr][0];
      const h8 lo = *(const h8*)(p + 8 * lg);
      const h8 hi = *(const h8*)(p + 16 + 8 * lg);
#pragma unroll
      for (int i = 0; i < 8; ++i) { afrag[tm][i] = lo[i]; afrag[tm][8 + i] = hi[i]; }
    }
    // ---- B fragments (32x16 f16): lane lr = col; ISA half-wave layout ----
#pragma unroll
    for (int tn = 0; tn < 4; ++tn) {
      const _Float16* p = &Ws[cur][wn * 64 + tn * 16 + lr][0];
      const h8 lo = *(const h8*)(p + 8 * lg);
      const h8 hi = *(const h8*)(p + 16 + 8 * lg);
#pragma unroll
      for (int i = 0; i < 8; ++i) { bfrag[tn][i] = lo[i]; bfrag[tn][8 + i] = hi[i]; }
    }

#pragma unroll
    for (int tm = 0; tm < 2; ++tm)
#pragma unroll
      for (int tn = 0; tn < 4; ++tn)
        acc[tm][tn] = __builtin_amdgcn_wmma_f32_16x16x32_f16(
            false, afrag[tm], false, bfrag[tn],
            (short)0, acc[tm][tn], false, false);

    // My async writes into `nxt` landed; barrier publishes them and retires
    // everyone's reads of `cur` before it is restaged next iteration.
    wait_async0();
    __syncthreads();
  }

  // ---- epilogue: C/D layout -> lane lr = col, VGPR v = row (v + 8*lg) ----
#pragma unroll
  for (int tm = 0; tm < 2; ++tm) {
#pragma unroll
    for (int tn = 0; tn < 4; ++tn) {
      const int col   = n0 + wn * 64 + tn * 16 + lr;
      const float bv  = bias[col];
      const int rbase = m0 + wm * 32 + tm * 16 + 8 * lg;
      if (mode == 0) {  // hidden layer: bias + relu, f16 out
        _Float16* O = (_Float16*)outp;
#pragma unroll
        for (int v = 0; v < 8; ++v) {
          float r = acc[tm][tn][v] + bv;
          O[(long)(rbase + v) * N + col] = (_Float16)(r > 0.f ? r : 0.f);
        }
      } else {          // final layer: bias + per-row validity mask, f32 out
        float* O = (float*)outp;
#pragma unroll
        for (int v = 0; v < 8; ++v) {
          const float r = acc[tm][tn][v] + bv;
          O[(long)(rbase + v) * N + col] = valid[rbase + v] ? r : 0.f;
        }
      }
    }
  }
}

// ---------------------------------------------------------------------------
extern "C" void kernel_launch(void* const* d_in, const int* in_sizes, int n_in,
                              void* d_out, int out_size, void* d_ws, size_t ws_size,
                              hipStream_t stream) {
  const float* A     = (const float*)d_in[0];  // [8192,512]
  const float* Bt    = (const float*)d_in[1];  // [100000,512]
  const int*   idx   = (const int*)d_in[2];    // [8192,16]
  const float* W_in  = (const float*)d_in[3];  // [1024,1024]
  const float* b_in  = (const float*)d_in[4];  // [1024]
  const float* W_h   = (const float*)d_in[5];  // [1024,1024]
  const float* b_h   = (const float*)d_in[6];  // [1024]
  const float* W_out = (const float*)d_in[7];  // [1024,512]
  const float* b_out = (const float*)d_in[8];  // [512]

  char* ws = (char*)d_ws;
  _Float16* WT_in = (_Float16*)(ws + (size_t)(0u  << 20));  // [1024][1024] f16
  _Float16* WT_h  = (_Float16*)(ws + (size_t)(2u  << 20));  // [1024][1024] f16
  _Float16* WT_o  = (_Float16*)(ws + (size_t)(4u  << 20));  // [512][1024]  f16
  _Float16* Xc    = (_Float16*)(ws + (size_t)(5u  << 20));  // [8192][1024] f16
  _Float16* H1    = (_Float16*)(ws + (size_t)(21u << 20));  // [8192][1024] f16
  _Float16* H2    = (_Float16*)(ws + (size_t)(37u << 20));  // [8192][1024] f16
  int*      valid = (int*)     (ws + (size_t)(53u << 20));  // [8192]

  const dim3 tb(32, 8);
  wt_transpose_f16<<<dim3(D_HID / 32, D_IN / 32), tb, 0, stream>>>(W_in, WT_in, D_IN, D_HID);
  wt_transpose_f16<<<dim3(D_HID / 32, D_HID / 32), tb, 0, stream>>>(W_h, WT_h, D_HID, D_HID);
  wt_transpose_f16<<<dim3(D_MODEL / 32, D_HID / 32), tb, 0, stream>>>(W_out, WT_o, D_HID, D_MODEL);

  gather_mean_pack<<<BATCH_A, 256, 0, stream>>>(A, Bt, idx, Xc, valid);

  gemm_wmma<<<dim3(BATCH_A / BM, D_HID / BN), 256, 0, stream>>>(
      Xc, WT_in, b_in, (void*)H1, nullptr, BATCH_A, D_HID, D_IN, 0);
  gemm_wmma<<<dim3(BATCH_A / BM, D_HID / BN), 256, 0, stream>>>(
      H1, WT_h, b_h, (void*)H2, nullptr, BATCH_A, D_HID, D_HID, 0);
  gemm_wmma<<<dim3(BATCH_A / BM, D_MODEL / BN), 256, 0, stream>>>(
      H2, WT_o, b_out, d_out, valid, BATCH_A, D_MODEL, D_HID, 1);
}
